// InnerAttention_45818711113843
// MI455X (gfx1250) — matrix operand order repaired and checked by reading, not compiled, not run on hardware
//
#include <hip/hip_runtime.h>

#define T_DIM 2048
#define S_DIM 2048
#define B_DIM 2
#define E_DIM 256
#define H_DIM 8
#define DH    32

typedef __attribute__((ext_vector_type(16))) __bf16 v16bf;
typedef __attribute__((ext_vector_type(8)))  __bf16 v8bf;
typedef __attribute__((ext_vector_type(8)))  float  v8f;
typedef __attribute__((ext_vector_type(4)))  float  v4f;

// hardware RTNE f32 -> bf16 (v_cvt, not emulated)
__device__ __forceinline__ __bf16 f2bf(float f) { return (__bf16)f; }

#define SHUF16(lo, hi) __builtin_shufflevector(lo, hi, 0, 1, 2, 3, 4, 5, 6, 7, 8, 9, 10, 11, 12, 13, 14, 15)
#define SHUF8(lo, hi)  __builtin_shufflevector(lo, hi, 0, 1, 2, 3, 4, 5, 6, 7)

// A-fragment (16x32 bf16, MxK) from row-major fp32, leading dim ld.
// Lane L: row = L&15, kb = 16*(L>=16); elems [i] = K(kb+i)
__device__ __forceinline__ v16bf load_a_f32(const float* p, int ld) {
    const int lane = threadIdx.x & 31;
    const float* r = p + (size_t)(lane & 15) * ld + ((lane >> 4) << 4);
    v4f x0 = *(const v4f*)(r);
    v4f x1 = *(const v4f*)(r + 4);
    v4f x2 = *(const v4f*)(r + 8);
    v4f x3 = *(const v4f*)(r + 12);
    v8bf lo = __builtin_convertvector(SHUF8(x0, x1), v8bf);
    v8bf hi = __builtin_convertvector(SHUF8(x2, x3), v8bf);
    return SHUF16(lo, hi);
}

__device__ __forceinline__ v16bf load_a_bf(const __bf16* p, int ld) {
    const int lane = threadIdx.x & 31;
    const __bf16* r = p + (size_t)(lane & 15) * ld + ((lane >> 4) << 4);
    v8bf lo = *(const v8bf*)(r);        // ds/global b128
    v8bf hi = *(const v8bf*)(r + 8);
    return SHUF16(lo, hi);
}

// B-fragment (32x16 bf16, KxN) from an N-major / K-contiguous source (src[n][k], stride ld).
// Lane L: col = L&15, kb = 16*(L>=16); elems [i] = B[kb+i][col]
__device__ __forceinline__ v16bf load_b_f32(const float* p, int ld) {
    const int lane = threadIdx.x & 31;
    const float* r = p + (size_t)(lane & 15) * ld + ((lane >> 4) << 4);
    v4f x0 = *(const v4f*)(r);
    v4f x1 = *(const v4f*)(r + 4);
    v4f x2 = *(const v4f*)(r + 8);
    v4f x3 = *(const v4f*)(r + 12);
    v8bf lo = __builtin_convertvector(SHUF8(x0, x1), v8bf);
    v8bf hi = __builtin_convertvector(SHUF8(x2, x3), v8bf);
    return SHUF16(lo, hi);
}

__device__ __forceinline__ v16bf load_b_bf(const __bf16* p, int ld) {
    const int lane = threadIdx.x & 31;
    const __bf16* r = p + (size_t)(lane & 15) * ld + ((lane >> 4) << 4);
    v8bf lo = *(const v8bf*)(r);
    v8bf hi = *(const v8bf*)(r + 8);
    return SHUF16(lo, hi);
}

// ---------------------------------------------------------------------------
// Kernel 1: Q/K projection. q[t,b,f] = (sum_e src[t,b,e]*W[f,e] + bias[f])*scale
// written bf16 in [b][h][t][d] layout. grid (T/16, E/64, 2*B), block 128.
// ---------------------------------------------------------------------------
__global__ void proj_qk_kernel(const float* __restrict__ query,
                               const float* __restrict__ key,
                               const float* __restrict__ Wq, const float* __restrict__ bq,
                               const float* __restrict__ Wk, const float* __restrict__ bk,
                               __bf16* __restrict__ q_ws, __bf16* __restrict__ k_ws) {
    const int t0   = blockIdx.x * 16;
    const int wave = threadIdx.x >> 5;
    const int f0   = (blockIdx.y * 4 + wave) * 16;
    const int b    = blockIdx.z & 1;
    const bool is_k = (blockIdx.z >> 1) != 0;

    const float* src  = is_k ? key : query;
    const float* W    = is_k ? Wk : Wq;
    const float* bias = is_k ? bk : bq;
    __bf16* dst       = is_k ? k_ws : q_ws;
    const float scale = is_k ? 1.0f : 0.17677669529663687f;  // 1/sqrt(Dh) folded into q

    const float* arow = src + (size_t)t0 * (B_DIM * E_DIM) + b * E_DIM;
    v8f c = {};
    for (int k0 = 0; k0 < E_DIM; k0 += 32) {
        v16bf a  = load_a_f32(arow + k0, B_DIM * E_DIM);
        v16bf bm = load_b_f32(W + (size_t)f0 * E_DIM + k0, E_DIM);
        c = __builtin_amdgcn_wmma_f32_16x16x32_bf16(false, a, false, bm, (short)0, c, false, false);
    }
    const int lane  = threadIdx.x & 31;
    const int col   = lane & 15;
    const int rbase = (lane >> 4) << 3;
    const int f = f0 + col;
    const int h = f >> 5, d = f & 31;
#pragma unroll
    for (int i = 0; i < 8; ++i) {
        int t = t0 + rbase + i;
        dst[((size_t)(b * H_DIM + h) * T_DIM + t) * DH + d] = f2bf((c[i] + bias[f]) * scale);
    }
}

// ---------------------------------------------------------------------------
// Kernel 2: V transpose/downconvert: value[s][b][h*32+d] -> v_ws[b][h][d][s] bf16.
// LDS tile transpose, coalesced on both sides. grid (S/64, B*H), block 256.
// ---------------------------------------------------------------------------
__global__ void vtrans_kernel(const float* __restrict__ value, __bf16* __restrict__ v_ws) {
    __shared__ float tile[64][DH + 1];
    const int bh = blockIdx.y;
    const int b = bh / H_DIM, h = bh % H_DIM;
    const int s0 = blockIdx.x * 64;
    for (int i = threadIdx.x; i < 64 * DH; i += 256) {
        int sl = i >> 5, d = i & 31;
        tile[sl][d] = value[(size_t)(s0 + sl) * (B_DIM * E_DIM) + b * E_DIM + h * DH + d];
    }
    __syncthreads();
    for (int i = threadIdx.x; i < 64 * DH; i += 256) {
        int d = i >> 6, sl = i & 63;
        v_ws[((size_t)bh * DH + d) * S_DIM + s0 + sl] = f2bf(tile[sl][d]);
    }
}

// ---------------------------------------------------------------------------
// Kernel 3: attention. One block = (b, 16 t-rows), loops over 8 heads.
// LDS: sc fp32 [16][S] 128KB | avg fp32 [16][S] 128KB | pb bf16 [16][S] 64KB
//      = 320KB (WGP max). oacc[16][32] aliases dead sc space during PV.
// grid (T/16, B), block 256 (8 waves).
// ---------------------------------------------------------------------------
__global__ void attn_kernel(const __bf16* __restrict__ q_ws,
                            const __bf16* __restrict__ k_ws,
                            const __bf16* __restrict__ v_ws,
                            float* __restrict__ att_out,    // [T,B,E]
                            float* __restrict__ avg_out) {  // [B,T,S]
    extern __shared__ char smem_raw[];
    float*  sc   = (float*)smem_raw;                     // [16][S]
    float*  avg  = (float*)(smem_raw + 16 * S_DIM * 4);  // [16][S]
    __bf16* pb   = (__bf16*)(smem_raw + 32 * S_DIM * 4); // [16][S]
    float*  oacc = sc;                                   // alias (sc dead during PV)

    const int t0   = blockIdx.x * 16;
    const int b    = blockIdx.y;
    const int tid  = threadIdx.x;
    const int lane = tid & 31;
    const int wave = tid >> 5;   // 0..7
    const int row  = tid >> 4;   // 0..15 softmax row
    const int sub  = tid & 15;   // 0..15 column slice

    for (int h = 0; h < H_DIM; ++h) {
        const __bf16* qb = q_ws + ((size_t)(b * H_DIM + h) * T_DIM + t0) * DH;
        const __bf16* kbp = k_ws + (size_t)(b * H_DIM + h) * S_DIM * DH;

        // ---- scores: 128 16x16 tiles, one 16x16x32 WMMA each (K = Dh = 32) ----
        v16bf aq = load_a_bf(qb, DH);
        for (int st = wave; st < S_DIM / 16; st += 8) {
            if (st + 8 < S_DIM / 16)
                __builtin_prefetch(kbp + (size_t)(st + 8) * 16 * DH, 0, 1);  // global_prefetch_b8
            v16bf bkf = load_b_bf(kbp + (size_t)st * 16 * DH, DH);
            v8f c = {};
            c = __builtin_amdgcn_wmma_f32_16x16x32_bf16(false, aq, false, bkf, (short)0, c, false, false);
            const int col   = lane & 15;
            const int rbase = (lane >> 4) << 3;
#pragma unroll
            for (int i = 0; i < 8; ++i)
                sc[(size_t)(rbase + i) * S_DIM + st * 16 + col] = c[i];
        }
        __syncthreads();

        // ---- softmax along S (16 threads per row, wave32 xor-shuffles) ----
        float mx = -3.4e38f;
        for (int s = sub; s < S_DIM; s += 16) mx = fmaxf(mx, sc[(size_t)row * S_DIM + s]);
#pragma unroll
        for (int m = 1; m < 16; m <<= 1) mx = fmaxf(mx, __shfl_xor(mx, m, 32));
        float sum = 0.f;
        for (int s = sub; s < S_DIM; s += 16) {
            float e = __expf(sc[(size_t)row * S_DIM + s] - mx);
            sc[(size_t)row * S_DIM + s] = e;
            sum += e;
        }
#pragma unroll
        for (int m = 1; m < 16; m <<= 1) sum += __shfl_xor(sum, m, 32);
        const float inv = 1.0f / sum;
        if (h == 0) {
            for (int s = sub; s < S_DIM; s += 16) {
                float p = sc[(size_t)row * S_DIM + s] * inv;
                pb[(size_t)row * S_DIM + s] = f2bf(p);
                avg[(size_t)row * S_DIM + s] = p;
            }
        } else {
            for (int s = sub; s < S_DIM; s += 16) {
                float p = sc[(size_t)row * S_DIM + s] * inv;
                pb[(size_t)row * S_DIM + s] = f2bf(p);
                avg[(size_t)row * S_DIM + s] += p;
            }
        }
        __syncthreads();   // pass3 done before oacc (aliases sc) is cleared

        for (int i = tid; i < 16 * DH; i += 256) oacc[i] = 0.f;
        __syncthreads();

        // ---- PV: out[16][32] = probs[16][S] x V[S][32]; waves split (d-tile, S-slice)
        const int dtile  = wave >> 2;  // d columns 0-15 / 16-31
        const int kslice = wave & 3;   // S partition
        const int dcol   = dtile * 16 + (lane & 15);
        const __bf16* vrow = v_ws + ((size_t)(b * H_DIM + h) * DH + dcol) * S_DIM;
        const __bf16* prow = pb + (size_t)(lane & 15) * S_DIM;
        v8f c = {};
        for (int sk = kslice; sk < S_DIM / 32; sk += 4) {
            const int s0 = sk * 32;
            if (sk + 4 < S_DIM / 32)
                __builtin_prefetch(vrow + s0 + 128, 0, 1);
            v16bf a;
            {   // probs bf16 from LDS: two ds_load_b128
                const __bf16* r = prow + s0 + ((lane >> 4) << 4);
                v8bf lo = *(const v8bf*)(r);
                v8bf hi = *(const v8bf*)(r + 8);
                a = SHUF16(lo, hi);
            }
            v16bf bv;
            {   // V^T bf16 from global: two b128 (contiguous in s)
                const __bf16* r = vrow + s0 + ((lane >> 4) << 4);
                v8bf lo = *(const v8bf*)(r);
                v8bf hi = *(const v8bf*)(r + 8);
                bv = SHUF16(lo, hi);
            }
            c = __builtin_amdgcn_wmma_f32_16x16x32_bf16(false, a, false, bv, (short)0, c, false, false);
        }
        {
            const int rbase = (lane >> 4) << 3;
#pragma unroll
            for (int i = 0; i < 8; ++i)
                atomicAdd(&oacc[(rbase + i) * DH + dcol], c[i]);  // ds_add_f32
        }
        __syncthreads();

        for (int i = tid; i < 16 * DH; i += 256) {
            int r = i >> 5, d = i & 31;
            att_out[(size_t)(t0 + r) * (B_DIM * E_DIM) + b * E_DIM + h * DH + d] = oacc[i];
        }
        __syncthreads();
    }

    // mean over heads, coalesced store (the bandwidth-critical output)
    const float invH = 1.0f / H_DIM;
    for (int s = sub; s < S_DIM; s += 16)
        avg_out[((size_t)b * T_DIM + t0 + row) * S_DIM + s] = avg[(size_t)row * S_DIM + s] * invH;
}

// ---------------------------------------------------------------------------
// Kernel 4: residual output projection: out = x + x@Wo^T + bo, x = att_out
// grid (T*B/16, E/64), block 128.
// ---------------------------------------------------------------------------
__global__ void oproj_kernel(const float* __restrict__ x,
                             const float* __restrict__ Wo, const float* __restrict__ bo,
                             float* __restrict__ out) {
    const int m0   = blockIdx.x * 16;
    const int wave = threadIdx.x >> 5;
    const int f0   = (blockIdx.y * 4 + wave) * 16;
    const float* arow = x + (size_t)m0 * E_DIM;
    v8f c = {};
    for (int k0 = 0; k0 < E_DIM; k0 += 32) {
        v16bf a  = load_a_f32(arow + k0, E_DIM);
        v16bf bm = load_b_f32(Wo + (size_t)f0 * E_DIM + k0, E_DIM);
        c = __builtin_amdgcn_wmma_f32_16x16x32_bf16(false, a, false, bm, (short)0, c, false, false);
    }
    const int lane  = threadIdx.x & 31;
    const int col   = lane & 15;
    const int rbase = (lane >> 4) << 3;
    const int f = f0 + col;
#pragma unroll
    for (int i = 0; i < 8; ++i) {
        int m = m0 + rbase + i;
        out[(size_t)m * E_DIM + f] = x[(size_t)m * E_DIM + f] + c[i] + bo[f];
    }
}

extern "C" void kernel_launch(void* const* d_in, const int* in_sizes, int n_in,
                              void* d_out, int out_size, void* d_ws, size_t ws_size,
                              hipStream_t stream) {
    const float* query = (const float*)d_in[0];
    const float* key   = (const float*)d_in[1];
    const float* value = (const float*)d_in[2];
    const float* Wq    = (const float*)d_in[3];
    const float* bq    = (const float*)d_in[4];
    const float* Wk    = (const float*)d_in[5];
    const float* bk    = (const float*)d_in[6];
    const float* Wo    = (const float*)d_in[7];
    const float* bo    = (const float*)d_in[8];

    // workspace: q_ws 2MB bf16 | k_ws 2MB bf16 | v_ws 2MB bf16 | att_out 4MB fp32
    __bf16* q_ws = (__bf16*)d_ws;
    __bf16* k_ws = q_ws + (size_t)B_DIM * H_DIM * T_DIM * DH;
    __bf16* v_ws = k_ws + (size_t)B_DIM * H_DIM * S_DIM * DH;
    float* att_out = (float*)((char*)d_ws + 6u * 1024u * 1024u);

    float* out = (float*)d_out;                          // [T,B,E]
    float* avg = out + (size_t)T_DIM * B_DIM * E_DIM;    // [B,T,S]

    proj_qk_kernel<<<dim3(T_DIM / 16, E_DIM / 64, 2 * B_DIM), 128, 0, stream>>>(
        query, key, Wq, bq, Wk, bk, q_ws, k_ws);

    vtrans_kernel<<<dim3(S_DIM / 64, B_DIM * H_DIM), 256, 0, stream>>>(value, v_ws);

    const size_t lds_bytes = (size_t)(32 * S_DIM) * sizeof(float)
                           + (size_t)(16 * S_DIM) * sizeof(__bf16);  // 327680 = 320KB
    attn_kernel<<<dim3(T_DIM / 16, B_DIM), 256, lds_bytes, stream>>>(
        q_ws, k_ws, v_ws, att_out, avg);

    oproj_kernel<<<dim3(T_DIM * B_DIM / 16, E_DIM / 64), 128, 0, stream>>>(
        att_out, Wo, bo, out);
}
